// CapsGNN_72069551227103
// MI455X (gfx1250) — compile-verified, matrix operand order checked by reading, not run on hardware
//
#include <hip/hip_runtime.h>
#include <hip/hip_bf16.h>
#include <math.h>

#define N_NODES 50000
#define F 128
#define T_OUT 10
#define MT 5              // M-tiles (of 16 rows) per wave; 625 blocks * 80 rows = 50000 exact

typedef __attribute__((ext_vector_type(16))) __bf16 v16bf;
typedef __attribute__((ext_vector_type(8)))  float  v8f;

__device__ __forceinline__ void atomicAddF32(float* p, float v) {
    __hip_atomic_fetch_add(p, v, __ATOMIC_RELAXED, __HIP_MEMORY_SCOPE_AGENT);
}

// ---------------- degree / norm precompute ----------------

__global__ void zero_f32_kernel(float* __restrict__ p, int n) {
    int i = blockIdx.x * blockDim.x + threadIdx.x;
    if (i < n) p[i] = 0.0f;
}

__global__ void deg_kernel(const long long* __restrict__ dst, int E,
                           float* __restrict__ deg) {
    int e = blockIdx.x * blockDim.x + threadIdx.x;
    int total = E + N_NODES;
    if (e >= total) return;
    int d = (e < E) ? (int)dst[e] : (e - E);
    atomicAddF32(&deg[d], 1.0f);
}

__global__ void dinv_kernel(float* __restrict__ deg, int n) {
    int i = blockIdx.x * blockDim.x + threadIdx.x;
    if (i >= n) return;
    float d = deg[i];
    deg[i] = (d > 0.0f) ? rsqrtf(d) : 0.0f;
}

__global__ void norm_kernel(const long long* __restrict__ src,
                            const long long* __restrict__ dst, int E,
                            const float* __restrict__ dinv,
                            float* __restrict__ norm) {
    int e = blockIdx.x * blockDim.x + threadIdx.x;
    int total = E + N_NODES;
    if (e >= total) return;
    int s, d;
    if (e < E) { s = (int)src[e]; d = (int)dst[e]; } else { s = d = e - E; }
    norm[e] = dinv[s] * dinv[d];
}

// ---------------- W -> bf16, packed in WMMA B-fragment order ----------------
// Element order: (((ntile*4 + kstep)*32 + lane)*16 + j), so each lane's 16-elem
// fragment is a contiguous 32B run -> two global_load_b128 per k-step in GEMM.
// B layout (16-bit, 32x16): lanes 0-15 -> N=lane, K in {kb..kb+7, kb+16..kb+23}
// (koff=0); lanes 16-31 -> N=lane-16, koff=8.  kb = kstep*32 + koff.

__global__ void pack_W_kernel(const float* __restrict__ W,
                              __bf16* __restrict__ Wp) {
    int i = blockIdx.x * blockDim.x + threadIdx.x;   // 16384 total
    if (i >= F * F) return;
    int j    = i & 15;
    int lane = (i >> 4) & 31;
    int ks   = (i >> 9) & 3;
    int nt   = i >> 11;
    int n  = nt * 16 + (lane & 15);
    int kb = ks * 32 + ((lane >> 4) << 3);
    int k  = (j < 8) ? (kb + j) : (kb + 16 + (j - 8));
    Wp[i] = (__bf16)W[k * F + n];
}

// ---------------- WMMA GEMM: H = X @ W  (M=50000, K=N=128) ----------------
// Block = 256 threads = 8 waves. Wave w owns cols [16w,16w+16) and MT=5
// stacked 16-row tiles: 20 x v_wmma_f32_16x16x32_bf16 per wave, one B
// fragment (2x b128 load) shared by 5 tiles per k-step.

__global__ void gcn_gemm_wmma_kernel(const float* __restrict__ X,
                                     const __bf16* __restrict__ Wp,
                                     float* __restrict__ H) {
    const int tid  = threadIdx.x;
    const int wave = tid >> 5;
    const int lane = tid & 31;
    const int half = lane >> 4;              // 0: lanes 0-15, 1: lanes 16-31
    const int koff = half << 3;              // K sub-offset 0 or 8
    const int lrow = lane & 15;
    const int row0 = blockIdx.x * (16 * MT);
    const int n    = wave * 16 + lrow;       // B/C/D column

    v8f c[MT];
    #pragma unroll
    for (int t = 0; t < MT; ++t) c[t] = (v8f){};

    #pragma unroll
    for (int ks = 0; ks < 4; ++ks) {
        // B fragment: contiguous 32B per lane from the packed buffer
        const v16bf b = *(const v16bf*)(Wp + (((size_t)wave * 4 + ks) * 32 + lane) * 16);
        const int kb = ks * 32 + koff;
        #pragma unroll
        for (int t = 0; t < MT; ++t) {
            const int row = row0 + t * 16 + lrow;
            const float* __restrict__ pa = X + (size_t)row * F + kb;
            const float4 q0 = *(const float4*)(pa);
            const float4 q1 = *(const float4*)(pa + 4);
            const float4 q2 = *(const float4*)(pa + 16);
            const float4 q3 = *(const float4*)(pa + 20);
            v16bf a;
            a[0]  = (__bf16)q0.x; a[1]  = (__bf16)q0.y;
            a[2]  = (__bf16)q0.z; a[3]  = (__bf16)q0.w;
            a[4]  = (__bf16)q1.x; a[5]  = (__bf16)q1.y;
            a[6]  = (__bf16)q1.z; a[7]  = (__bf16)q1.w;
            a[8]  = (__bf16)q2.x; a[9]  = (__bf16)q2.y;
            a[10] = (__bf16)q2.z; a[11] = (__bf16)q2.w;
            a[12] = (__bf16)q3.x; a[13] = (__bf16)q3.y;
            a[14] = (__bf16)q3.z; a[15] = (__bf16)q3.w;
            c[t] = __builtin_amdgcn_wmma_f32_16x16x32_bf16(
                       /*neg_a=*/false, a, /*neg_b=*/false, b,
                       /*c_mod=*/(short)0, c[t],
                       /*reuse_a=*/false, /*reuse_b=*/false);
        }
    }

    // C/D layout: VGPR r -> M = r + 8*half, N = lane&15
    #pragma unroll
    for (int t = 0; t < MT; ++t) {
        #pragma unroll
        for (int r = 0; r < 8; ++r) {
            int m = row0 + t * 16 + r + (half << 3);
            H[(size_t)m * F + n] = c[t][r];
        }
    }
}

// ---------------- out = b (broadcast), then edge scatter-add ----------------

__global__ void init_bias_kernel(const float* __restrict__ b,
                                 float* __restrict__ out) {
    int i = blockIdx.x * blockDim.x + threadIdx.x;
    if (i < N_NODES * F) out[i] = b[i & (F - 1)];
}

// One wave32 per edge; lane covers 4 contiguous floats (float4 gather,
// 4x global_atomic_add_f32 scatter). h/out are L2-resident (25.6 MB each).
__global__ void agg_kernel(const float* __restrict__ H,
                           const long long* __restrict__ src,
                           const long long* __restrict__ dst,
                           const float* __restrict__ norm,
                           float* __restrict__ out, int E) {
    int gid  = blockIdx.x * blockDim.x + threadIdx.x;
    int e    = gid >> 5;
    int lane = gid & 31;
    int total = E + N_NODES;
    if (e >= total) return;
    int s, d;
    if (e < E) { s = (int)src[e]; d = (int)dst[e]; } else { s = d = e - E; }
    float nv = norm[e];
    const float4 v = *(const float4*)(H + (size_t)s * F + lane * 4);
    float* po = out + (size_t)d * F + lane * 4;
    atomicAddF32(po + 0, v.x * nv);
    atomicAddF32(po + 1, v.y * nv);
    atomicAddF32(po + 2, v.z * nv);
    atomicAddF32(po + 3, v.w * nv);
}

// ---------------- mean pool + head + log_softmax ----------------

__global__ void pool_kernel(const float* __restrict__ x,
                            float* __restrict__ pooled) {
    int col = threadIdx.x;  // 128 threads: coalesced column access
    float s = 0.0f;
    for (int r = blockIdx.x; r < N_NODES; r += gridDim.x)
        s += x[(size_t)r * F + col];
    atomicAddF32(&pooled[col], s);
}

__global__ void head_kernel(const float* __restrict__ pooled,
                            const float* __restrict__ Wout,
                            const float* __restrict__ bout,
                            float* __restrict__ out) {
    __shared__ float sl[T_OUT];
    __shared__ float red[2];
    int t = threadIdx.x;
    if (t < T_OUT) {
        float acc = bout[t];
        const float inv_n = 1.0f / (float)N_NODES;
        for (int f = 0; f < F; ++f)
            acc += (pooled[f] * inv_n) * Wout[f * T_OUT + t];
        sl[t] = acc;
    }
    __syncthreads();
    if (t == 0) {
        float m = -1e30f;
        for (int i = 0; i < T_OUT; ++i) m = fmaxf(m, sl[i]);
        float s = 0.0f;
        for (int i = 0; i < T_OUT; ++i) s += __expf(sl[i] - m);
        red[0] = m;
        red[1] = logf(s);
    }
    __syncthreads();
    if (t < T_OUT) out[t] = sl[t] - red[0] - red[1];
}

// ---------------- driver ----------------

extern "C" void kernel_launch(void* const* d_in, const int* in_sizes, int n_in,
                              void* d_out, int out_size, void* d_ws, size_t ws_size,
                              hipStream_t stream) {
    const float*     feat  = (const float*)d_in[0];
    const long long* edges = (const long long*)d_in[1];   // int64 [2, E] flat
    const float*     W0    = (const float*)d_in[2];
    const float*     b0    = (const float*)d_in[3];
    const float*     W1    = (const float*)d_in[4];
    const float*     b1    = (const float*)d_in[5];
    const float*     W2    = (const float*)d_in[6];
    const float*     b2    = (const float*)d_in[7];
    const float*     Wout  = (const float*)d_in[8];
    const float*     bout  = (const float*)d_in[9];
    float* out = (float*)d_out;

    const int E     = in_sizes[1] / 2;
    const int total = E + N_NODES;
    const long long* esrc = edges;
    const long long* edst = edges + E;

    // Workspace carve-up (256B-aligned slices)
    char*  ws  = (char*)d_ws;
    size_t off = 0;
    auto carveb = [&](size_t nbytes) -> char* {
        char* p = ws + off;
        off += ((nbytes + 255) / 256) * 256;
        return p;
    };
    float*  deg    = (float*)carveb((size_t)N_NODES * 4);   // becomes dinv in-place
    float*  norm   = (float*)carveb((size_t)total * 4);
    float*  h      = (float*)carveb((size_t)N_NODES * F * 4);
    float*  buf0   = (float*)carveb((size_t)N_NODES * F * 4);
    float*  buf1   = (float*)carveb((size_t)N_NODES * F * 4);
    float*  pooled = (float*)carveb((size_t)F * 4);
    __bf16* Wp     = (__bf16*)carveb((size_t)F * F * 2);    // packed bf16 weights

    // degree -> dinv -> per-edge norm
    zero_f32_kernel<<<(N_NODES + 255) / 256, 256, 0, stream>>>(deg, N_NODES);
    deg_kernel<<<(total + 255) / 256, 256, 0, stream>>>(edst, E, deg);
    dinv_kernel<<<(N_NODES + 255) / 256, 256, 0, stream>>>(deg, N_NODES);
    norm_kernel<<<(total + 255) / 256, 256, 0, stream>>>(esrc, edst, E, deg, norm);

    // Three GCN layers: pack W -> GEMM (WMMA) -> init with bias -> scatter-add
    struct Layer { const float* x; const float* Wm; const float* bv; float* dst; };
    Layer L[3] = { {feat, W0, b0, buf0},
                   {buf0, W1, b1, buf1},
                   {buf1, W2, b2, buf0} };
    const int gemm_blocks = N_NODES / (16 * MT);                   // 625 exact
    const int agg_blocks  = (int)(((long long)total * 32 + 255) / 256);
    for (int l = 0; l < 3; ++l) {
        pack_W_kernel<<<(F * F + 255) / 256, 256, 0, stream>>>(L[l].Wm, Wp);
        gcn_gemm_wmma_kernel<<<gemm_blocks, 256, 0, stream>>>(L[l].x, Wp, h);
        init_bias_kernel<<<(N_NODES * F + 255) / 256, 256, 0, stream>>>(L[l].bv, L[l].dst);
        agg_kernel<<<agg_blocks, 256, 0, stream>>>(h, esrc, edst, norm, L[l].dst, E);
    }

    // mean pool + head + log_softmax
    zero_f32_kernel<<<1, 128, 0, stream>>>(pooled, F);
    pool_kernel<<<512, 128, 0, stream>>>(buf0, pooled);
    head_kernel<<<1, 32, 0, stream>>>(pooled, Wout, bout, out);
}
